// MaxPoolIGMC_89807766159943
// MI455X (gfx1250) — matrix operand order repaired and checked by reading, not compiled, not run on hardware
//
#include <hip/hip_runtime.h>

// MI455X / gfx1250: wave32, WMMA f32_16x16x32_f16 for all matrix work.
// All hot buffers (<50MB total) are L2-resident (192MB); the bound is L2
// atomic/scatter bandwidth (E*32 f32 adds per layer), so: native f32 atomics
// (unsafeAtomicAdd -> global_atomic_add_f32) + prefetch of streaming edge
// metadata. WMMA FLOPs (even with 5x relation redundancy) are negligible.

typedef __attribute__((ext_vector_type(16))) _Float16 v16h;
typedef __attribute__((ext_vector_type(8)))  _Float16 v8h;
typedef __attribute__((ext_vector_type(8)))  float    v8f;

#define NREL 5

union V16 { v16h v; v8h h[2]; };

__device__ __forceinline__ v8f wmma_f16(v16h a, v16h b, v8f c) {
  // (neg_a, A, neg_b, B, c_mod, C, reuse_a, reuse_b)
  return __builtin_amdgcn_wmma_f32_16x16x32_f16(false, a, false, b, (short)0, c, false, false);
}

__device__ __forceinline__ void atomic_add_f32(float* p, float v) {
  unsafeAtomicAdd(p, v);  // native global_atomic_add_f32 (no CAS loop)
}

// ---------------- prep kernels ----------------

// h16[node][col] = col<4 ? x[node][col] : 0   (zero-pad so WMMA K=32 is exact)
__global__ void init_h16(const float* __restrict__ x, _Float16* __restrict__ h16, int N) {
  int i = blockIdx.x * blockDim.x + threadIdx.x;
  if (i >= N * 32) return;
  int node = i >> 5, col = i & 31;
  h16[i] = (col < 4) ? (_Float16)x[node * 4 + col] : (_Float16)0.f;
}

// W[l][r] = comp[l] @ basis[l]  (zero-padded to 32x32), written pre-swizzled
// into the per-lane B-operand layout this code uses (contiguous k, matching A):
//   lane L = 16*(k>>4) + (n&15), half h = k&15, col-tile j = n>>4
__global__ void pack_rel_weights(const float* b0, const float* c0, const float* b1, const float* c1,
                                 const float* b2, const float* c2, const float* b3, const float* c3,
                                 _Float16* __restrict__ bpack) {
  int idx = blockIdx.x * blockDim.x + threadIdx.x;
  if (idx >= 4 * NREL * 32 * 32) return;
  int n = idx & 31, k = (idx >> 5) & 31;
  int r = (idx >> 10) % NREL;
  int l = idx / (NREL * 1024);
  const float* ba; const float* co; int in;
  switch (l) {
    case 0: ba = b0; co = c0; in = 4;  break;
    case 1: ba = b1; co = c1; in = 32; break;
    case 2: ba = b2; co = c2; in = 32; break;
    default: ba = b3; co = c3; in = 32; break;
  }
  float w = 0.f;
  if (k < in)
    w = co[r * 2 + 0] * ba[0 * in * 32 + k * 32 + n] +
        co[r * 2 + 1] * ba[1 * in * 32 + k * 32 + n];
  int L = ((k >> 4) << 4) + (n & 15);
  int h = k & 15;
  int j = n >> 4;
  bpack[(((l * NREL + r) * 2 + j) * 32 + L) * 16 + h] = (_Float16)w;
}

// Same swizzle for the per-layer root matrices.
__global__ void pack_root(const float* r0, const float* r1, const float* r2, const float* r3,
                          _Float16* __restrict__ rpack) {
  int idx = blockIdx.x * blockDim.x + threadIdx.x;
  if (idx >= 4 * 32 * 32) return;
  int n = idx & 31, k = (idx >> 5) & 31;
  int l = idx >> 10;
  const float* ro; int in;
  switch (l) {
    case 0: ro = r0; in = 4;  break;
    case 1: ro = r1; in = 32; break;
    case 2: ro = r2; in = 32; break;
    default: ro = r3; in = 32; break;
  }
  float w = (k < in) ? ro[k * 32 + n] : 0.f;
  int L = ((k >> 4) << 4) + (n & 15);
  int h = k & 15;
  int j = n >> 4;
  rpack[((l * 2 + j) * 32 + L) * 16 + h] = (_Float16)w;
}

// cnt[dst][etype]++ ; then per-edge scale = 1/max(cnt,1). Counts are
// layer-invariant so this runs once per launch.
__global__ void count_edges(const int* __restrict__ dst, const int* __restrict__ et,
                            int* __restrict__ cnt, int E) {
  int e = blockIdx.x * blockDim.x + threadIdx.x;
  if (e < E) atomicAdd(&cnt[dst[e] * NREL + et[e]], 1);
}

__global__ void edge_scale(const int* __restrict__ dst, const int* __restrict__ et,
                           const int* __restrict__ cnt, float* __restrict__ esc, int E) {
  int e = blockIdx.x * blockDim.x + threadIdx.x;
  if (e >= E) return;
  int c = cnt[dst[e] * NREL + et[e]];
  esc[e] = 1.0f / (float)(c < 1 ? 1 : c);
}

// ---------------- edge message kernel (WMMA) ----------------
// One wave per 16-edge tile. A = h16[src] rows (f16, 16x32 A in contiguous k order k = 16g+s).
// 5 relations x 2 column-halves of chained WMMAs on one accumulator; rows
// not belonging to relation r are zeroed in A (register select), so the
// accumulation over relations is exact and EXEC stays all-ones for WMMA.
// Scatter scaled by 1/cnt (mean aggregation) with native f32 atomics.
__global__ void __launch_bounds__(256) edge_msg(
    const _Float16* __restrict__ h16, const int* __restrict__ src,
    const int* __restrict__ dst, const int* __restrict__ et,
    const float* __restrict__ esc, const _Float16* __restrict__ bpack,
    float* __restrict__ agg, int E) {
  int lane = threadIdx.x & 31;
  int wave = threadIdx.x >> 5;
  int m = lane & 15, g = lane >> 4;

  // Hoist all 5 relations' B tiles into registers (layer weights are constant).
  V16 B[NREL][2];
#pragma unroll
  for (int r = 0; r < NREL; r++)
#pragma unroll
    for (int j = 0; j < 2; j++) {
      const _Float16* p = bpack + (((r * 2 + j) * 32) + lane) * 16;
      B[r][j].h[0] = *(const v8h*)p;
      B[r][j].h[1] = *(const v8h*)(p + 8);
    }
  v16h zero = {};

  int tiles = (E + 15) >> 4;
  int nwaves = (gridDim.x * blockDim.x) >> 5;
  for (int t = blockIdx.x * (blockDim.x >> 5) + wave; t < tiles; t += nwaves) {
    int base = t << 4;

    // Prefetch next grid-stride tile's streaming edge metadata into cache
    // (global_prefetch_b8; no counter cost) while this tile computes.
    int tn = t + nwaves;
    if (tn < tiles) {
      int nb = (tn << 4) + m;
      __builtin_prefetch(&src[nb], 0, 0);
      __builtin_prefetch(&dst[nb], 0, 0);
      __builtin_prefetch(&et[nb], 0, 0);
      __builtin_prefetch(&esc[nb], 0, 0);
    }

    int eidx = base + m;
    bool valid = eidx < E;
    int ec = valid ? eidx : (E - 1);
    int s    = src[ec];
    int myet = et[ec];
    int d    = dst[ec];
    float sc = valid ? esc[ec] : 0.f;

    // A operand: lane (m,g) holds K = 16g..16g+15 of row m.
    const _Float16* hp = h16 + (size_t)s * 32;
    V16 A;
    A.h[0] = *(const v8h*)(hp + 16 * g);
    A.h[1] = *(const v8h*)(hp + 16 * g + 8);

    v8f c0 = {}, c1 = {};
#pragma unroll
    for (int r = 0; r < NREL; r++) {
      v16h ar = (myet == r) ? A.v : zero;     // zero non-matching rows
      c0 = wmma_f16(ar, B[r][0].v, c0);       // out cols 0..15
      c1 = wmma_f16(ar, B[r][1].v, c1);       // out cols 16..31
    }

    // D layout: VGPR v of lane (m,g) holds element (row 8g+v, col m).
#pragma unroll
    for (int v = 0; v < 8; v++) {
      int row = v + 8 * g;
      int dd  = __shfl(d, row, 32);
      float s2 = __shfl(sc, row, 32);
      if (s2 != 0.f) {
        float* ap = agg + (size_t)dd * 32 + m;
        atomic_add_f32(ap,      c0[v] * s2);
        atomic_add_f32(ap + 16, c1[v] * s2);
      }
    }
  }
}

// ---------------- node finalize (WMMA root + bias + agg + tanh + maxpool) ----
__global__ void __launch_bounds__(256) node_finalize(
    const _Float16* __restrict__ h16, const _Float16* __restrict__ rpack,
    const float* __restrict__ bias, const float* __restrict__ agg,
    float* __restrict__ mp, _Float16* __restrict__ h16out, int N, int first) {
  int lane = threadIdx.x & 31;
  int wave = threadIdx.x >> 5;
  int m = lane & 15, g = lane >> 4;

  V16 B0, B1;
  {
    const _Float16* p = rpack + lane * 16;
    B0.h[0] = *(const v8h*)p;  B0.h[1] = *(const v8h*)(p + 8);
    p = rpack + (32 + lane) * 16;
    B1.h[0] = *(const v8h*)p;  B1.h[1] = *(const v8h*)(p + 8);
  }
  float bsl = bias[m], bsh = bias[m + 16];

  int tiles = N >> 4;  // N divisible by 16
  int nwaves = (gridDim.x * blockDim.x) >> 5;
  for (int t = blockIdx.x * (blockDim.x >> 5) + wave; t < tiles; t += nwaves) {
    int base = t << 4;
    const _Float16* hp = h16 + (size_t)(base + m) * 32;
    V16 A;
    A.h[0] = *(const v8h*)(hp + 16 * g);
    A.h[1] = *(const v8h*)(hp + 16 * g + 8);
    v8f c0 = {}, c1 = {};
    c0 = wmma_f16(A.v, B0.v, c0);
    c1 = wmma_f16(A.v, B1.v, c1);
#pragma unroll
    for (int v = 0; v < 8; v++) {
      int node = base + v + 8 * g;
      size_t o = (size_t)node * 32 + m;
      float t0 = tanhf(c0[v] + bsl + agg[o]);
      float t1 = tanhf(c1[v] + bsh + agg[o + 16]);
      if (first) { mp[o] = t0; mp[o + 16] = t1; }
      else       { mp[o] = fmaxf(mp[o], t0); mp[o + 16] = fmaxf(mp[o + 16], t1); }
      h16out[o]      = (_Float16)t0;
      h16out[o + 16] = (_Float16)t1;
    }
  }
}

// ---------------- MLP head: [B,64] -> relu(128) -> 1 ----------------
__global__ void __launch_bounds__(128) head_mlp(
    const float* __restrict__ mp, const int* __restrict__ uidx, const int* __restrict__ iidx,
    const float* __restrict__ w1, const float* __restrict__ b1,
    const float* __restrict__ w2, const float* __restrict__ b2, float* __restrict__ out) {
  __shared__ float feat[64];
  __shared__ float ssum[4];
  int b = blockIdx.x;
  int tid = threadIdx.x;
  if (tid < 64) {
    int node = (tid < 32) ? uidx[b] : iidx[b];
    feat[tid] = mp[(size_t)node * 32 + (tid & 31)];
  }
  __syncthreads();
  float acc = b1[tid];
#pragma unroll
  for (int k = 0; k < 64; k++) acc += feat[k] * w1[k * 128 + tid];
  acc = fmaxf(acc, 0.f) * w2[tid];
  for (int off = 16; off > 0; off >>= 1) acc += __shfl_down(acc, off, 32);
  if ((tid & 31) == 0) ssum[tid >> 5] = acc;
  __syncthreads();
  if (tid == 0) out[b] = ssum[0] + ssum[1] + ssum[2] + ssum[3] + b2[0];
}

// ---------------- host ----------------
extern "C" void kernel_launch(void* const* d_in, const int* in_sizes, int n_in,
                              void* d_out, int out_size, void* d_ws, size_t ws_size,
                              hipStream_t stream) {
  const float* x  = (const float*)d_in[0];
  const int* esrc = (const int*)d_in[1];
  const int* edst = (const int*)d_in[2];
  const int* eet  = (const int*)d_in[3];
  const int* uidx = (const int*)d_in[4];
  const int* iidx = (const int*)d_in[5];
  const float *basis[4], *comp[4], *root[4], *bias[4];
  for (int l = 0; l < 4; l++) {
    basis[l] = (const float*)d_in[6 + 4 * l];
    comp[l]  = (const float*)d_in[7 + 4 * l];
    root[l]  = (const float*)d_in[8 + 4 * l];
    bias[l]  = (const float*)d_in[9 + 4 * l];
  }
  const float* w1 = (const float*)d_in[22];
  const float* b1 = (const float*)d_in[23];
  const float* w2 = (const float*)d_in[24];
  const float* b2 = (const float*)d_in[25];
  float* out = (float*)d_out;

  const int N = in_sizes[0] / 4;
  const int E = in_sizes[1];
  const int B = in_sizes[4];

  // workspace carve-out (~48 MB)
  char* wsb = (char*)d_ws;
  size_t off = 0;
  auto carve = [&](size_t bytes) -> void* {
    void* p = wsb + off;
    off += (bytes + 255) & ~(size_t)255;
    return p;
  };
  _Float16* h16a  = (_Float16*)carve((size_t)N * 32 * 2);
  _Float16* h16b  = (_Float16*)carve((size_t)N * 32 * 2);
  float*    mp    = (float*)carve((size_t)N * 32 * 4);
  float*    agg   = (float*)carve((size_t)N * 32 * 4);
  int*      cnt   = (int*)carve((size_t)N * NREL * 4);
  float*    esc   = (float*)carve((size_t)E * 4);
  _Float16* bpack = (_Float16*)carve((size_t)4 * NREL * 2 * 32 * 16 * 2);
  _Float16* rpack = (_Float16*)carve((size_t)4 * 2 * 32 * 16 * 2);

  hipMemsetAsync(cnt, 0, (size_t)N * NREL * 4, stream);
  init_h16<<<(N * 32 + 255) / 256, 256, 0, stream>>>(x, h16a, N);
  pack_rel_weights<<<(4 * NREL * 1024 + 255) / 256, 256, 0, stream>>>(
      basis[0], comp[0], basis[1], comp[1], basis[2], comp[2], basis[3], comp[3], bpack);
  pack_root<<<(4 * 1024 + 255) / 256, 256, 0, stream>>>(root[0], root[1], root[2], root[3], rpack);
  count_edges<<<(E + 255) / 256, 256, 0, stream>>>(edst, eet, cnt, E);
  edge_scale<<<(E + 255) / 256, 256, 0, stream>>>(edst, eet, cnt, esc, E);

  _Float16* hc = h16a;
  _Float16* hn = h16b;
  int node_tiles = N / 16;
  int fin_blocks = (node_tiles + 7) / 8;
  for (int l = 0; l < 4; l++) {
    hipMemsetAsync(agg, 0, (size_t)N * 32 * 4, stream);
    edge_msg<<<1024, 256, 0, stream>>>(hc, esrc, edst, eet, esc,
                                       bpack + (size_t)l * NREL * 2 * 32 * 16, agg, E);
    node_finalize<<<fin_blocks, 256, 0, stream>>>(hc, rpack + (size_t)l * 2 * 32 * 16,
                                                  bias[l], agg, mp, hn, N, (l == 0) ? 1 : 0);
    _Float16* tsw = hc; hc = hn; hn = tsw;
  }
  head_mlp<<<B, 128, 0, stream>>>(mp, uidx, iidx, w1, b1, w2, b2, out);
}
